// attention_85830626443604
// MI455X (gfx1250) — matrix operand order repaired and checked by reading, not compiled, not run on hardware
//
#include <hip/hip_runtime.h>

// ---------------------------------------------------------------------------
// softmax-free causal attention for MI455X (gfx1250, wave32, WMMA bf16)
//   Q = e@Wq^T + bq ; V = e@Wv^T + bv ; out = tril(Q @ P^T) @ V
// B=8, T=2048, E=512, fp32 in/out; internal bf16 operands, f32 accumulation.
//
// Pass 0: one-shot f32->bf16 conversion of e, P, Wq, Wv (pure bandwidth).
// Pass 1: fused Q/V projection, pure load_b128 -> v_wmma inner loop.
// Pass 2: streaming causal attention; Q tile staged in LDS via
//         GLOBAL_LOAD_ASYNC_TO_LDS_B128, scores shared through LDS.
// ---------------------------------------------------------------------------

#define BB 8
#define TT 2048
#define EE 512

typedef __attribute__((ext_vector_type(16))) __bf16 v16bf;
typedef __attribute__((ext_vector_type(8)))  __bf16 v8bf;
typedef __attribute__((ext_vector_type(4)))  __bf16 v4bf;
typedef __attribute__((ext_vector_type(8)))  float  v8f;

__device__ __forceinline__ __bf16 f2bf(float f) {
  unsigned u = __builtin_bit_cast(unsigned, f);
  u += 0x7FFFu + ((u >> 16) & 1u);              // round-to-nearest-even
  unsigned short s = (unsigned short)(u >> 16);
  return __builtin_bit_cast(__bf16, s);
}

#define CAT16(lo, hi) __builtin_shufflevector((lo), (hi), 0,1,2,3,4,5,6,7,8,9,10,11,12,13,14,15)

#define WMMA_BF16(a, b, c) \
  __builtin_amdgcn_wmma_f32_16x16x32_bf16(false, (a), false, (b), (short)0, (c), false, false)

// gfx1250 async copy: global -> LDS, tracked by ASYNCcnt (no VGPR round-trip).
// VDST carries the LDS byte offset (low 32 bits of the flat shared address).
__device__ __forceinline__ void async_load_b128(unsigned ldsOff, const void* gptr) {
  asm volatile("global_load_async_to_lds_b128 %0, %1, off"
               :: "v"(ldsOff), "v"(gptr) : "memory");
}
__device__ __forceinline__ void wait_asynccnt0() {
  asm volatile("s_wait_asynccnt 0x0" ::: "memory");
}

// ---------------------------------------------------------------------------
// Kernel 0: generic f32 -> bf16 conversion (each element converted ONCE)
// ---------------------------------------------------------------------------
__global__ __launch_bounds__(256) void cvt_kernel(const float* __restrict__ src,
                                                  __bf16* __restrict__ dst, int n4) {
  for (int i = blockIdx.x * blockDim.x + threadIdx.x; i < n4;
       i += gridDim.x * blockDim.x) {
    float4 f = ((const float4*)src)[i];
    v4bf o;
    o[0] = f2bf(f.x); o[1] = f2bf(f.y); o[2] = f2bf(f.z); o[3] = f2bf(f.w);
    ((v4bf*)dst)[i] = o;
  }
}

// ---------------------------------------------------------------------------
// Kernel 1: fused Q/V projection. One wave -> one 16x16 tile of Q and of V
// (shared A-fragment of e). All operands pre-converted bf16: the k-loop is
// pure b128 loads + v_wmma. Q stored bf16 [B*T,E]; V stored TRANSPOSED
// bf16 [B,E,T] so the scores@V GEMM gets contiguous B-fragments.
// ---------------------------------------------------------------------------
__global__ __launch_bounds__(256) void qv_proj_kernel(
    const __bf16* __restrict__ Ebf, const __bf16* __restrict__ Wqbf,
    const float* __restrict__ bq,   const __bf16* __restrict__ Wvbf,
    const float* __restrict__ bv,
    __bf16* __restrict__ Qbf, __bf16* __restrict__ Vt) {
  const int lane = threadIdx.x & 31;
  const int wave = threadIdx.x >> 5;
  const int gid  = blockIdx.x * 8 + wave;       // (B*T/16) * (E/16) tiles
  const int mt   = gid >> 5;                    // row tile in [0, B*T/16)
  const int ft   = gid & 31;                    // col tile in [0, E/16)
  const int m0   = mt * 16;
  const int f0   = ft * 16;
  const int ln16 = lane & 15;
  const int half = lane >> 4;

  const __bf16* aBase = Ebf  + (size_t)(m0 + ln16) * EE;   // A: row m, contiguous K
  const __bf16* qBase = Wqbf + (size_t)(f0 + ln16) * EE;   // B: col n = W row, contiguous K
  const __bf16* vBase = Wvbf + (size_t)(f0 + ln16) * EE;

  v8f qacc = {};
  v8f vacc = {};
  for (int kk = 0; kk < EE; kk += 32) {
    v8bf lo = *(const v8bf*)(aBase + kk + half * 16);
    v8bf hi = *(const v8bf*)(aBase + kk + half * 16 + 8);
    v16bf a    = CAT16(lo, hi);
    v16bf bq16 = *(const v16bf*)(qBase + kk + half * 16);
    v16bf bv16 = *(const v16bf*)(vBase + kk + half * 16);
    qacc = WMMA_BF16(a, bq16, qacc);
    vacc = WMMA_BF16(a, bv16, vacc);
  }

  const float bqn = bq[f0 + ln16];
  const float bvn = bv[f0 + ln16];
  const int   b   = m0 >> 11;                   // / T
  const int   tb  = m0 & (TT - 1);
#pragma unroll
  for (int r = 0; r < 8; ++r) {
    const int m = r + 8 * half;                 // C/D layout: lane n=L%16, row r+8*(L/16)
    Qbf[(size_t)(m0 + m) * EE + f0 + ln16] = f2bf(qacc[r] + bqn);
    Vt[((size_t)b * EE + f0 + ln16) * TT + (tb + m)] = f2bf(vacc[r] + bvn);
  }
}

// ---------------------------------------------------------------------------
// Kernel 2: streaming causal attention. Block = 8 waves = one 16-row t-tile.
// The block's 16x512 Q tile is staged ONCE into LDS with async b128 copies
// (all 8 waves previously re-read it from global every s-block). Then per
// 256-wide s-block: each wave computes a 32-wide scores chunk (masked,
// bf16 -> LDS), barrier, then each wave accumulates its 64-wide e-slice of
// out += S_tile @ V^T. Linear accumulation (no softmax) => no rescaling.
// ---------------------------------------------------------------------------
__global__ __launch_bounds__(256) void attn_kernel(
    const __bf16* __restrict__ Qbf, const __bf16* __restrict__ Pbf,
    const __bf16* __restrict__ Vt, float* __restrict__ out) {
  // padded row strides: 264*2B=528B and 520*2B=1040B are both == 4 DWORDs
  // (mod 64 banks) per +1 row -> 16-row b128 fragment reads are conflict-free
  __shared__ __align__(32) __bf16 S[16][264];   // scores stage (8.25 KB)
  __shared__ __align__(32) __bf16 Qs[16][520];  // Q tile stage (16.25 KB)

  const int lane = threadIdx.x & 31;
  const int wave = threadIdx.x >> 5;
  const int ln16 = lane & 15;
  const int half = lane >> 4;
  const int b    = blockIdx.x >> 7;             // T/16 = 128 tiles per batch
  const int t0   = (blockIdx.x & 127) * 16;

  // ---- stage Q[t0..t0+15][0..511] into LDS via async copy engine ----
  {
    const __bf16* Qg = Qbf + (size_t)(b * TT + t0) * EE;
    for (int i = threadIdx.x; i < 1024; i += 256) {   // 1024 x 16B chunks
      const int r = i >> 6;                           // row (64 chunks/row)
      const int c = (i & 63) * 8;                     // bf16 column
      async_load_b128((unsigned)(uintptr_t)&Qs[r][c], Qg + (size_t)r * EE + c);
    }
    wait_asynccnt0();
  }
  __syncthreads();

  v8f acc[4] = {{}, {}, {}, {}};
  const int ec0 = wave * 64;                    // this wave's e-slice

  const __bf16* Vbase = Vt + ((size_t)b * EE + ec0 + ln16) * TT + half * 16;
  const int tEnd = t0 + 16;

  for (int sb = 0; sb < tEnd; sb += 256) {
    // ---- Phase A: scores for s-chunk [sb + wave*32, +32), Q from LDS ----
    const int sc = sb + wave * 32;
    v8f s0acc = {};
    v8f s1acc = {};
    {
      const __bf16* P0 = Pbf + (size_t)(b * TT + sc + ln16) * EE + half * 16;
      const __bf16* P1 = P0 + (size_t)16 * EE;
      for (int kk = 0; kk < EE; kk += 32) {
        v8bf lo = *(const v8bf*)(&Qs[ln16][kk + half * 16]);
        v8bf hi = *(const v8bf*)(&Qs[ln16][kk + half * 16 + 8]);
        v16bf aQ = CAT16(lo, hi);
        v16bf b0 = *(const v16bf*)(P0 + kk);
        v16bf b1 = *(const v16bf*)(P1 + kk);
        s0acc = WMMA_BF16(aQ, b0, s0acc);
        s1acc = WMMA_BF16(aQ, b1, s1acc);
      }
    }
    // causal mask (keep s <= t) applied during f32->bf16, then stage in LDS
#pragma unroll
    for (int r = 0; r < 8; ++r) {
      const int m = r + 8 * half;
      const int t = t0 + m;
      const int s = sc + ln16;
      S[m][wave * 32 + ln16]      = f2bf(s      <= t ? s0acc[r] : 0.0f);
      S[m][wave * 32 + 16 + ln16] = f2bf(s + 16 <= t ? s1acc[r] : 0.0f);
    }
    __syncthreads();

    // prefetch next s-block's V^T slice while we crunch this one
#pragma unroll
    for (int nt = 0; nt < 4; ++nt) {
      __builtin_prefetch(Vbase + (size_t)nt * 16 * TT + sb + 256, 0, 1);
    }

    // ---- Phase B: out += S(16x256) @ V^T slice (256 x 64) ----
    for (int koff = 0; koff < 256; koff += 32) {
      v8bf lo = *(const v8bf*)(&S[ln16][koff + half * 16]);
      v8bf hi = *(const v8bf*)(&S[ln16][koff + half * 16 + 8]);
      v16bf aS = CAT16(lo, hi);
#pragma unroll
      for (int nt = 0; nt < 4; ++nt) {
        v16bf bV = *(const v16bf*)(Vbase + (size_t)nt * 16 * TT + sb + koff);
        acc[nt] = WMMA_BF16(aS, bV, acc[nt]);
      }
    }
    __syncthreads();
  }

  // ---- store fp32 output [B,T,E] ----
#pragma unroll
  for (int nt = 0; nt < 4; ++nt) {
    const int col = ec0 + nt * 16 + ln16;
#pragma unroll
    for (int r = 0; r < 8; ++r) {
      const int t = t0 + r + 8 * half;
      out[(size_t)(b * TT + t) * EE + col] = acc[nt][r];
    }
  }
}

// ---------------------------------------------------------------------------
extern "C" void kernel_launch(void* const* d_in, const int* in_sizes, int n_in,
                              void* d_out, int out_size, void* d_ws, size_t ws_size,
                              hipStream_t stream) {
  (void)in_sizes; (void)n_in; (void)out_size; (void)ws_size;

  const float* e  = (const float*)d_in[0];
  const float* pl = (const float*)d_in[1];
  const float* Wq = (const float*)d_in[2];
  const float* bq = (const float*)d_in[3];
  const float* Wv = (const float*)d_in[4];
  const float* bv = (const float*)d_in[5];
  float* out = (float*)d_out;

  const size_t N  = (size_t)BB * TT * EE;       // 8M elements
  const size_t NW = (size_t)EE * EE;            // 256K elements
  __bf16* Ebf  = (__bf16*)d_ws;                 // 16 MB
  __bf16* Pbf  = Ebf + N;                       // 16 MB
  __bf16* Qbf  = Pbf + N;                       // 16 MB
  __bf16* Vt   = Qbf + N;                       // 16 MB (transposed V: [B,E,T])
  __bf16* Wqbf = Vt + N;                        // 0.5 MB
  __bf16* Wvbf = Wqbf + NW;                     // 0.5 MB

  cvt_kernel<<<2048, 256, 0, stream>>>(e,  Ebf,  (int)(N / 4));
  cvt_kernel<<<2048, 256, 0, stream>>>(pl, Pbf,  (int)(N / 4));
  cvt_kernel<<<256,  256, 0, stream>>>(Wq, Wqbf, (int)(NW / 4));
  cvt_kernel<<<256,  256, 0, stream>>>(Wv, Wvbf, (int)(NW / 4));
  // (B*T/16)*(E/16) = 32768 tiles, 8 waves per 256-thread block -> 4096 blocks
  qv_proj_kernel<<<4096, 256, 0, stream>>>(Ebf, Wqbf, bq, Wvbf, bv, Qbf, Vt);
  // one block per (b, 16-row t-tile): 8 * 128 = 1024 blocks
  attn_kernel<<<BB * (TT / 16), 256, 0, stream>>>(Qbf, Pbf, Vt, out);
}
